// CtxAtt_21827023798504
// MI455X (gfx1250) — matrix operand order repaired and checked by reading, not compiled, not run on hardware
//
#include <hip/hip_runtime.h>
#include <hip/hip_bf16.h>
#include <hip/hip_fp16.h>

typedef __attribute__((ext_vector_type(16))) _Float16 v16h;
typedef __attribute__((ext_vector_type(8)))  float    v8f;

#define TOK    4096   // R*C tokens per batch
#define DMODEL 512
#define NHEAD  8
#define HDIM   64
#define NBATCH 2

// ---- CDNA5 async copy: global -> LDS, 16B per lane, tracked by ASYNCcnt ----
__device__ __forceinline__ void async_ld_b128(const void* gsrc, void* lds_dst) {
  const unsigned lofs = (unsigned)(unsigned long long)lds_dst;  // LDS aperture low bits
  asm volatile("global_load_async_to_lds_b128 %0, %1, off"
               :: "v"(lofs), "v"(gsrc) : "memory");
}
__device__ __forceinline__ void wait_async0() {
  asm volatile("s_wait_asynccnt 0" ::: "memory");
}

// ---- fragment loaders (gfx1250 wave32 WMMA, contiguous k order: lane half h holds k 16h..16h+15 in A and B) ----

// A: 16x32 f16, row-major source in LDS with row stride `ld` (halves).
// lanes 0-15: M=lane, K 0..15
// lanes 16-31: M=lane-16, K 16..31
__device__ __forceinline__ v16h load_a_frag(const _Float16* __restrict__ lds,
                                            int ld, int m0, int k0, int lane) {
  const int m  = m0 + (lane & 15);
  const int kb = k0 + ((lane & 16) ? 16 : 0);
  const _Float16* p = lds + m * ld + kb;
  v16h a;
#pragma unroll
  for (int e = 0; e < 8; ++e) a[e] = p[e];
#pragma unroll
  for (int e = 0; e < 8; ++e) a[8 + e] = p[8 + e];
  return a;
}

// B: 32x16 f16 from LDS stored k-major per column: bt[n*ld + k]
// lanes 0-15: N=lane, K=0..15 ; lanes 16-31: K=16..31  -> one contiguous 32B read
__device__ __forceinline__ v16h load_b_frag(const _Float16* __restrict__ bt,
                                            int ld, int n0, int k0, int lane) {
  const int n  = n0 + (lane & 15);
  const int kb = k0 + ((lane & 16) ? 16 : 0);
  const _Float16* p = bt + n * ld + kb;
  v16h b;
#pragma unroll
  for (int e = 0; e < 16; ++e) b[e] = p[e];
  return b;
}

__device__ __forceinline__ v8f wmma_f16(v16h a, v16h b, v8f c) {
  return __builtin_amdgcn_wmma_f32_16x16x32_f16(false, a, false, b, (short)0, c,
                                                false, false);
}

// ---- kernel 1: maskw[b,j] = mask[b,j] / (eps + sum_j mask[b,j]) ----
__global__ __launch_bounds__(256) void mask_kernel(const int* __restrict__ mask,
                                                   float* __restrict__ maskw) {
  __shared__ float red[256];
  const int b = blockIdx.x, tid = threadIdx.x;
  float s = 0.f;
  for (int i = tid; i < TOK; i += 256) s += (mask[b * TOK + i] != 0) ? 1.0f : 0.0f;
  red[tid] = s;
  __syncthreads();
  for (int off = 128; off > 0; off >>= 1) {
    if (tid < off) red[tid] += red[tid + off];
    __syncthreads();
  }
  const float inv = 1.0f / (1e-6f + red[0]);
  for (int i = tid; i < TOK; i += 256)
    maskw[b * TOK + i] = (mask[b * TOK + i] != 0) ? inv : 0.0f;
}

// ---- kernel 2: projections  Y = f16(X @ W + b), X f32 (8192x512), z picks q/k/v ----
__global__ __launch_bounds__(256) void proj_kernel(
    const float* __restrict__ qf, const float* __restrict__ kf, const float* __restrict__ vf,
    const float* __restrict__ Wq, const float* __restrict__ bq,
    const float* __restrict__ Wk, const float* __restrict__ bk,
    const float* __restrict__ Wv, const float* __restrict__ bv,
    _Float16* __restrict__ Qh, _Float16* __restrict__ Kh, _Float16* __restrict__ Vh) {
  const float* X; const float* W; const float* bias; _Float16* Y;
  if (blockIdx.z == 0)      { X = qf; W = Wq; bias = bq; Y = Qh; }
  else if (blockIdx.z == 1) { X = kf; W = Wk; bias = bk; Y = Kh; }
  else                      { X = vf; W = Wv; bias = bv; Y = Vh; }

  __shared__ _Float16 As[64 * 40];   // 64 rows x 32 k (padded, 80B rows)
  __shared__ _Float16 Bt[64 * 40];   // 64 cols x 32 k (k-major)

  const int r0 = blockIdx.y * 64, c0 = blockIdx.x * 64;
  const int tid = threadIdx.x, wave = tid >> 5, lane = tid & 31;
  const int mi = (wave & 3) * 16, ni0 = (wave >> 2) * 32;

  v8f acc0 = {}; v8f acc1 = {};

  for (int kk = 0; kk < DMODEL; kk += 32) {
    __syncthreads();
    {   // stage A (f32 -> f16), coalesced rows
      const int row = tid >> 2, seg = (tid & 3) * 8;
      const float* src = X + (size_t)(r0 + row) * DMODEL + kk + seg;
#pragma unroll
      for (int i = 0; i < 8; ++i) As[row * 40 + seg + i] = (_Float16)src[i];
    }
    {   // stage B transposed: W[kk+k][c0+n] -> Bt[n][k]
      const int krow = tid >> 3, seg = (tid & 7) * 8;
      const float* src = W + (size_t)(kk + krow) * DMODEL + c0 + seg;
#pragma unroll
      for (int i = 0; i < 8; ++i) Bt[(seg + i) * 40 + krow] = (_Float16)src[i];
    }
    __syncthreads();

    v16h a  = load_a_frag(As, 40, mi, 0, lane);
    v16h b0 = load_b_frag(Bt, 40, ni0, 0, lane);
    v16h b1 = load_b_frag(Bt, 40, ni0 + 16, 0, lane);
    acc0 = wmma_f16(a, b0, acc0);
    acc1 = wmma_f16(a, b1, acc1);
  }

  const int col0 = c0 + ni0 + (lane & 15);
  const int rowb = r0 + mi + ((lane & 16) ? 8 : 0);
  const float bb0 = bias[col0], bb1 = bias[col0 + 16];
#pragma unroll
  for (int j = 0; j < 8; ++j) {
    Y[(size_t)(rowb + j) * DMODEL + col0]      = (_Float16)(acc0[j] + bb0);
    Y[(size_t)(rowb + j) * DMODEL + col0 + 16] = (_Float16)(acc1[j] + bb1);
  }
}

// ---- kernel 3: flash-style sigmoid attention per (b, h, 64-query tile) ----
__global__ __launch_bounds__(256) void attn_kernel(
    const _Float16* __restrict__ Qh, const _Float16* __restrict__ Kh,
    const _Float16* __restrict__ Vh, const float* __restrict__ maskw,
    _Float16* __restrict__ Xa) {
  __shared__ _Float16 Qs[64 * 72];   // [query][d]   (144B rows, 16B aligned chunks)
  __shared__ _Float16 Ks[64 * 72];   // [key][d]     (B-frag friendly for Q.K^T)
  __shared__ _Float16 Vt[64 * 72];   // [d][key]     (B-frag friendly for P.V)
  __shared__ _Float16 Ps[64 * 72];   // [query][key] (A-frag friendly)

  const int qt = blockIdx.x, h = blockIdx.y, b = blockIdx.z;
  const int tid = threadIdx.x, wave = tid >> 5, lane = tid & 31;
  const int mi = (wave & 3) * 16, ni0 = (wave >> 2) * 32;
  const int ln = lane & 15;

  const size_t base = (size_t)b * TOK * DMODEL + (size_t)h * HDIM;

  // stage Q tile once: 512 x 16B chunks, async DMA to LDS (2 per thread)
  {
#pragma unroll
    for (int c = 0; c < 2; ++c) {
      const int chunk = tid + c * 256;          // 0..511
      const int row = chunk >> 3, seg = (chunk & 7) * 8;
      async_ld_b128(Qh + base + (size_t)(qt * 64 + row) * DMODEL + seg,
                    &Qs[row * 72 + seg]);
    }
  }

  v8f o0 = {}; v8f o1 = {};

  for (int kt = 0; kt < TOK / 64; ++kt) {
    __syncthreads();   // previous iteration's WMMA reads of Ks/Vt/Ps are done
    // stage K tile via async DMA (row-major, 2 x 16B chunks per thread)
#pragma unroll
    for (int c = 0; c < 2; ++c) {
      const int chunk = tid + c * 256;
      const int row = chunk >> 3, seg = (chunk & 7) * 8;
      async_ld_b128(Kh + base + (size_t)(kt * 64 + row) * DMODEL + seg,
                    &Ks[row * 72 + seg]);
    }
    {   // stage V transposed (needs lane shuffle -> synchronous path)
      const int row = tid >> 2, seg = (tid & 3) * 16;
      const _Float16* vs = Vh + base + (size_t)(kt * 64 + row) * DMODEL + seg;
#pragma unroll
      for (int i = 0; i < 16; ++i) Vt[(seg + i) * 72 + row] = vs[i];
    }
    wait_async0();     // this wave's async copies (K, and Q on iter 0) landed
    __syncthreads();

    // S = Q . K^T  (K-dim = d = 64 -> two 16x16x32 WMMAs per subtile)
    v8f s0 = {}; v8f s1 = {};
#pragma unroll
    for (int kc = 0; kc < HDIM; kc += 32) {
      v16h a  = load_a_frag(Qs, 72, mi, kc, lane);
      v16h b0 = load_b_frag(Ks, 72, ni0, kc, lane);
      v16h b1 = load_b_frag(Ks, 72, ni0 + 16, kc, lane);
      s0 = wmma_f16(a, b0, s0);
      s1 = wmma_f16(a, b1, s1);
    }

    // P = sigmoid(S/8) * mask/denom ; stage to LDS (C-layout -> A-layout)
    const int rowb = mi + ((lane & 16) ? 8 : 0);
    const float mw0 = maskw[b * TOK + kt * 64 + ni0 + ln];
    const float mw1 = maskw[b * TOK + kt * 64 + ni0 + 16 + ln];
#pragma unroll
    for (int j = 0; j < 8; ++j) {
      const float p0 = mw0 / (1.0f + __expf(-0.125f * s0[j]));
      const float p1 = mw1 / (1.0f + __expf(-0.125f * s1[j]));
      Ps[(rowb + j) * 72 + ni0 + ln]      = (_Float16)p0;
      Ps[(rowb + j) * 72 + ni0 + 16 + ln] = (_Float16)p1;
    }
    __syncthreads();

    // O += P . V  (K-dim = 64 keys)
#pragma unroll
    for (int kc = 0; kc < 64; kc += 32) {
      v16h a  = load_a_frag(Ps, 72, mi, kc, lane);
      v16h b0 = load_b_frag(Vt, 72, ni0, kc, lane);
      v16h b1 = load_b_frag(Vt, 72, ni0 + 16, kc, lane);
      o0 = wmma_f16(a, b0, o0);
      o1 = wmma_f16(a, b1, o1);
    }
  }

  const int rowb = qt * 64 + mi + ((lane & 16) ? 8 : 0);
#pragma unroll
  for (int j = 0; j < 8; ++j) {
    Xa[base + (size_t)(rowb + j) * DMODEL + ni0 + ln]      = (_Float16)o0[j];
    Xa[base + (size_t)(rowb + j) * DMODEL + ni0 + 16 + ln] = (_Float16)o1[j];
  }
}

// ---- kernel 4: out = Xa @ Wo + bo + shortcut(q), f32 out ----
__global__ __launch_bounds__(256) void out_kernel(
    const _Float16* __restrict__ Xa, const float* __restrict__ Wo,
    const float* __restrict__ bo, const float* __restrict__ qin,
    float* __restrict__ out) {
  __shared__ _Float16 As[64 * 40];
  __shared__ _Float16 Bt[64 * 40];

  const int r0 = blockIdx.y * 64, c0 = blockIdx.x * 64;
  const int tid = threadIdx.x, wave = tid >> 5, lane = tid & 31;
  const int mi = (wave & 3) * 16, ni0 = (wave >> 2) * 32;

  v8f acc0 = {}; v8f acc1 = {};

  for (int kk = 0; kk < DMODEL; kk += 32) {
    __syncthreads();
    {   // stage A (f16 row-major) via async DMA: 256 x 16B chunks, 1 per thread
      const int row = tid >> 2, seg = (tid & 3) * 8;
      async_ld_b128(Xa + (size_t)(r0 + row) * DMODEL + kk + seg,
                    &As[row * 40 + seg]);
    }
    {   // stage B transposed: Wo[kk+k][c0+n] -> Bt[n][k] (f32->f16, sync path)
      const int krow = tid >> 3, seg = (tid & 7) * 8;
      const float* src = Wo + (size_t)(kk + krow) * DMODEL + c0 + seg;
#pragma unroll
      for (int i = 0; i < 8; ++i) Bt[(seg + i) * 40 + krow] = (_Float16)src[i];
    }
    wait_async0();
    __syncthreads();

    v16h a  = load_a_frag(As, 40, mi, 0, lane);
    v16h b0 = load_b_frag(Bt, 40, ni0, 0, lane);
    v16h b1 = load_b_frag(Bt, 40, ni0 + 16, 0, lane);
    acc0 = wmma_f16(a, b0, acc0);
    acc1 = wmma_f16(a, b1, acc1);
  }

  const int col0 = c0 + ni0 + (lane & 15);
  const int rowb = r0 + mi + ((lane & 16) ? 8 : 0);
  const float bb0 = bo[col0], bb1 = bo[col0 + 16];
#pragma unroll
  for (int j = 0; j < 8; ++j) {
    const size_t idx = (size_t)(rowb + j) * DMODEL;
    out[idx + col0]      = acc0[j] + bb0 + qin[idx + col0];
    out[idx + col0 + 16] = acc1[j] + bb1 + qin[idx + col0 + 16];
  }
}

extern "C" void kernel_launch(void* const* d_in, const int* in_sizes, int n_in,
                              void* d_out, int out_size, void* d_ws, size_t ws_size,
                              hipStream_t stream) {
  (void)in_sizes; (void)n_in; (void)out_size; (void)ws_size;
  const float* q    = (const float*)d_in[0];
  const float* k    = (const float*)d_in[1];
  const float* v    = (const float*)d_in[2];
  const int*   mask = (const int*)d_in[3];
  const float* Wq   = (const float*)d_in[4];
  const float* bq   = (const float*)d_in[5];
  const float* Wk   = (const float*)d_in[6];
  const float* bk   = (const float*)d_in[7];
  const float* Wv   = (const float*)d_in[8];
  const float* bv   = (const float*)d_in[9];
  const float* Wo   = (const float*)d_in[10];
  const float* bo   = (const float*)d_in[11];
  float* out = (float*)d_out;

  const size_t mat = (size_t)NBATCH * TOK * DMODEL;   // 4,194,304 f16 elems
  _Float16* Qh = (_Float16*)d_ws;
  _Float16* Kh = Qh + mat;
  _Float16* Vh = Kh + mat;
  _Float16* Xa = Vh + mat;
  float* maskw = (float*)(Xa + mat);                  // NBATCH*TOK f32

  mask_kernel<<<NBATCH, 256, 0, stream>>>(mask, maskw);

  proj_kernel<<<dim3(DMODEL / 64, (NBATCH * TOK) / 64, 3), 256, 0, stream>>>(
      q, k, v, Wq, bq, Wk, bk, Wv, bv, Qh, Kh, Vh);

  attn_kernel<<<dim3(TOK / 64, NHEAD, NBATCH), 256, 0, stream>>>(
      Qh, Kh, Vh, maskw, Xa);

  out_kernel<<<dim3(DMODEL / 64, (NBATCH * TOK) / 64), 256, 0, stream>>>(
      Xa, Wo, bo, q, out);
}
